// ECConvLayer_38620345926111
// MI455X (gfx1250) — matrix operand order repaired and checked by reading, not compiled, not run on hardware
//
#include <hip/hip_runtime.h>

// ---------------------------------------------------------------------------
// ECConv fused GNN layer for MI455X (gfx1250, wave32, WMMA + TDM).
//
// Dominant cost: edge MLP  relu(E(65536x32) @ W_e(32x4096) + b_e)  = 17.2 GFLOP.
// Path: f16 V_WMMA_F32_16X16X32_F16 with f32 accumulate, bias folded into the
// WMMA C operand, transposed tiling so the d-contraction with h_src is
// per-lane v_med3(relu) + dual-issue FMA (co-executes with WMMA) + one
// shfl_xor(16). W_e tiles double-buffered into LDS with the Tensor Data Mover
// (TENSORcnt, in-order completion). Segment mean via L2-resident f32 atomics;
// final node GEMM also WMMA f16.
// ---------------------------------------------------------------------------

typedef __attribute__((ext_vector_type(16))) _Float16 v16h;
typedef __attribute__((ext_vector_type(8)))  float    v8f;
typedef __attribute__((ext_vector_type(4)))  unsigned v4u;
typedef __attribute__((ext_vector_type(8)))  int      v8i;
typedef __attribute__((ext_vector_type(4)))  int      v4i;

union U16h { float4 q[2]; v16h h; };
union U8f  { float4 q[2]; v8f  v; };

#define N_SRC   32768
#define N_DSTC  8192
#define NEDGE   65536
#define NODE_IN 64
#define EDGE_IN 32
#define HIDDEN  64
#define WCOLS   4096   // HIDDEN*NODE_IN

#define HAVE_TDM (__has_builtin(__builtin_amdgcn_tensor_load_to_lds) && \
                  __has_builtin(__builtin_amdgcn_s_wait_tensorcnt))

__device__ __forceinline__ float relu1(float v) {  // single v_med3_f32
  return __builtin_amdgcn_fmed3f(v, 0.0f, __builtin_inff());
}

#if HAVE_TDM
// One TDM descriptor moving 32 KB global -> LDS (1-D copy as data_size=8B,
// tensor_dim0 = tile_dim0 = 4096 elements, stride 4096, type=2, count=1).
__device__ __forceinline__ void tdm_copy32k(const void* gsrc, void* ldsDst) {
  unsigned ldsOff = (unsigned)(unsigned long long)ldsDst;   // flat->LDS low 32b
  unsigned long long ga = (unsigned long long)gsrc;
  v4u g0 = { 1u,                                            // count=1, user mode
             ldsOff,                                        // lds_addr
             (unsigned)(ga & 0xFFFFFFFFu),                  // global_addr[31:0]
             (unsigned)((ga >> 32) & 0x1FFFFFFu) | 0x80000000u }; // [56:32]|type=2
  v8i g1 = { 0x00030000,                                    // data_size=3 (8B)
             0x10000000,                                    // tensor_dim0=4096
             0x00010000,                                    // tensor_dim1=1
             0x10000000,                                    // tile_dim0=4096
             0x00000001,                                    // tile_dim1=1
             4096, 0, 0 };                                  // dim0_stride=4096
  v4i g2 = { 1, 1, 4096, 0 };                               // dim2/3=1, dim2_stride
  v4i g3 = { 4096, 1 << 16, 0, 0 };                         // dim3_stride, dim4=1
  v8i g4 = { 0, 0, 0, 0, 0, 0, 0, 0 };                      // extra group (zeroed)
  __builtin_amdgcn_tensor_load_to_lds(g0, g1, g2, g3, g4, 0);
}
#endif

// ---------------------------------------------------------------------------
// Prep: swizzle W_e (32x4096 f32) into f16 WMMA A-operand tiles (transposed:
// A row = W_e column), and W_n (128x64 f32) into f16 B-operand tiles.
// A-layout (16-bit, 16x32): lane row M = lane%16; element i holds
//   K = 8*(lane>=16) + (i<8 ? i : i+8).
// B-layout (16-bit, 32x16): lane col N = lane%16; element i holds
//   K = 8*(lane>=16) + (i<8 ? i : i+8).
// ---------------------------------------------------------------------------
__global__ __launch_bounds__(256) void prep_kernel(
    const float* __restrict__ We, const float* __restrict__ Wn,
    _Float16* __restrict__ wswz, _Float16* __restrict__ wnswz)
{
  int tid = blockIdx.x * 256 + threadIdx.x;
  if (tid < 256 * 32) {                       // W_e: 256 column-tiles of 16
    int cb = tid >> 5, lane = tid & 31;
    int col = cb * 16 + (lane & 15);          // A row = W_e column j
    int kb  = (lane >> 4) * 8;
    #pragma unroll
    for (int i = 0; i < 16; ++i) {
      int K = kb + (i < 8 ? i : i + 8);
      wswz[(size_t)(cb * 32 + lane) * 16 + i] = (_Float16)We[(size_t)K * WCOLS + col];
    }
  } else if (tid < 256 * 32 + 16 * 32) {      // W_n: 16 tiles (nb 0..3, kstep 0..3)
    int t = tid - 256 * 32;
    int tile = t >> 5, lane = t & 31;
    int nb = tile >> 2, s = tile & 3;
    int col = nb * 16 + (lane & 15);
    int kb  = s * 32 + (lane >> 4) * 8;
    #pragma unroll
    for (int i = 0; i < 16; ++i)
      wnswz[(size_t)(tile * 32 + lane) * 16 + i] = (_Float16)Wn[(size_t)(kb + (i < 8 ? i : i + 8)) * HIDDEN + col];
  }
}

__global__ __launch_bounds__(256) void zero_kernel(float* __restrict__ msum,
                                                   unsigned* __restrict__ cnt)
{
  int tid = blockIdx.x * 256 + threadIdx.x;
  if (tid < N_DSTC * HIDDEN) msum[tid] = 0.0f;
  if (tid < N_DSTC)          cnt[tid]  = 0u;
}

// ---------------------------------------------------------------------------
// Edge kernel: 8 waves/block, 16 edges/wave (128 edges/block, 512 blocks).
// Per wave: B-operand = its 16 edges' features (held in regs whole kernel);
// loop over 256 W-column tiles: D(16j x 16e) = W_tile^T x E^T + b_e, then
// acc += relu(D)*h_src per lane, fold halves with one shfl_xor(16), atomic.
// W_e tiles double-buffered through LDS in 32KB chunks via the TDM.
// ---------------------------------------------------------------------------
__global__ __launch_bounds__(256) void edge_kernel(
    const float* __restrict__ nf, const float* __restrict__ ef,
    const int* __restrict__ src, const int* __restrict__ dst,
    const _Float16* __restrict__ wswz, const float* __restrict__ be,
    float* __restrict__ msum, unsigned* __restrict__ cnt)
{
  __shared__ __align__(32) _Float16 lds_w[2 * 32 * 32 * 16]; // 2x32KB W chunks
  __shared__ __align__(16) float    lds_hs[8 * 16 * 64];     // 32KB gathered h_src

  const int lane = threadIdx.x & 31;
  const int wave = threadIdx.x >> 5;
  const int eRow = lane & 15;     // edge within tile (B column / D column)
  const int hi   = lane >> 4;     // lane-half selector
  const int e0   = (blockIdx.x * 8 + wave) * 16;

  // B operand: 16 edges x K=32, f16.  element i <-> K = 8*hi + (i<8 ? i : i+8).
  v16h bmat;
  {
    const float* er = ef + (size_t)(e0 + eRow) * EDGE_IN;
    float tmp[16];
    #pragma unroll
    for (int q = 0; q < 4; ++q) {
      float4 f = ((const float4*)er)[hi * 2 + (q < 2 ? q : q + 2)];
      tmp[q*4+0] = f.x; tmp[q*4+1] = f.y; tmp[q*4+2] = f.z; tmp[q*4+3] = f.w;
    }
    #pragma unroll
    for (int i = 0; i < 16; ++i) bmat[i] = (_Float16)tmp[i];
  }

  int sIdx = 0, dIdx = 0;
  if (lane < 16) { sIdx = src[e0 + lane]; dIdx = dst[e0 + lane]; }

#if HAVE_TDM
  if (wave == 0) tdm_copy32k(wswz, lds_w);    // prefetch chunk 0 ASAP
#endif

  // Gather the 16 h_src rows (16x64 f32) coalesced into this wave's LDS slab.
  float* hsw = &lds_hs[wave * 1024];
  #pragma unroll 4
  for (int j = 0; j < 32; ++j) {
    int s = __shfl(sIdx, j >> 1, 32);           // row index, uniform per iter
    int t = j * 32 + lane;                      // t = el*64 + dd, dd contiguous
    hsw[t] = nf[(size_t)s * NODE_IN + (t & 63)];
  }
  if (lane < 16) atomicAdd(&cnt[dIdx], 1u);     // segment counts

  __syncthreads();

  // h_src values this lane needs: hs[e=eRow][d = db*16 + 8*hi + r]
  float hsreg[32];
  #pragma unroll
  for (int db = 0; db < 4; ++db) {
    const float4* p = (const float4*)(hsw + eRow * 64 + db * 16 + hi * 8);
    float4 f0 = p[0], f1 = p[1];
    hsreg[db*8+0]=f0.x; hsreg[db*8+1]=f0.y; hsreg[db*8+2]=f0.z; hsreg[db*8+3]=f0.w;
    hsreg[db*8+4]=f1.x; hsreg[db*8+5]=f1.y; hsreg[db*8+6]=f1.z; hsreg[db*8+7]=f1.w;
  }

  for (int hc = 0; hc < 8; ++hc) {              // 8 chunks x (8 h-values x 4 d-blocks)
#if HAVE_TDM
    // Double-buffered TDM: issue next chunk, then wait for the current one.
    // TENSORcnt completes in-order per wave, so tensorcnt<=1 => chunk hc landed.
    if (wave == 0) {
      if (hc < 7) {
        tdm_copy32k((const char*)wswz + (size_t)(hc + 1) * 32768,
                    lds_w + (size_t)((hc + 1) & 1) * (32 * 32 * 16));
        __builtin_amdgcn_s_wait_tensorcnt(1);
      } else {
        __builtin_amdgcn_s_wait_tensorcnt(0);
      }
    }
    __syncthreads();                            // chunk hc visible to all waves
#else
    __syncthreads();                            // previous chunk fully consumed
    { // fallback: cooperative stage, 32 tiles = 32 KB = 2048 float4
      const float4* s4 = (const float4*)wswz + hc * 2048;
      float4* d4 = (float4*)(lds_w + (size_t)(hc & 1) * (32 * 32 * 16));
      for (int t = threadIdx.x; t < 2048; t += 256) d4[t] = s4[t];
    }
    __syncthreads();
#endif
    const _Float16* wbuf = lds_w + (size_t)(hc & 1) * (32 * 32 * 16);

    for (int hl = 0; hl < 8; ++hl) {
      const int h = hc * 8 + hl;
      float acc0 = 0.0f, acc1 = 0.0f;           // two chains -> dual-issue fmac
      #pragma unroll
      for (int db = 0; db < 4; ++db) {
        const int cbL = hl * 4 + db;
        U16h a;          // A operand: W_e^T tile from LDS (2x b128)
        const float4* ap = (const float4*)(wbuf + (size_t)(cbL * 32 + lane) * 16);
        a.q[0] = ap[0]; a.q[1] = ap[1];
        U8f c;           // C operand = bias b_e[j], j = row of transposed tile
        const float4* bp = (const float4*)(be + (size_t)(hc * 32 + cbL) * 16 + hi * 8);
        c.q[0] = bp[0]; c.q[1] = bp[1];
        v8f d = __builtin_amdgcn_wmma_f32_16x16x32_f16(
            false, a.h, false, bmat, (short)0, c.v, false, false);
        #pragma unroll
        for (int r = 0; r < 4; ++r) {           // relu (v_med3) + contract
          acc0 = fmaf(relu1(d[2*r+0]), hsreg[db*8 + 2*r+0], acc0);
          acc1 = fmaf(relu1(d[2*r+1]), hsreg[db*8 + 2*r+1], acc1);
        }
      }
      float acc = acc0 + acc1;
      // halves hold d%16 in [0,8) and [8,16): one xor-16 fold completes sum_d
      float m = acc + __shfl_xor(acc, 16, 32);
      if (lane < 16) atomicAdd(&msum[(size_t)dIdx * HIDDEN + h], m);
    }
    __syncthreads();                            // buffer free for TDM overwrite
  }
}

// ---------------------------------------------------------------------------
// Node kernel: out = relu([h_self | msum/cnt] @ W_n + b_n), 16 nodes/wave,
// K=128 as 4 f16 WMMA steps, b_n folded into C operand.
// ---------------------------------------------------------------------------
__global__ __launch_bounds__(256) void node_kernel(
    const float* __restrict__ nf, const float* __restrict__ msum,
    const unsigned* __restrict__ cnt, const _Float16* __restrict__ wnswz,
    const float* __restrict__ bn, float* __restrict__ out)
{
  const int lane = threadIdx.x & 31;
  const int wave = threadIdx.x >> 5;
  const int nRow = lane & 15;
  const int hi   = lane >> 4;
  const int n0   = (blockIdx.x * 8 + wave) * 16;
  const int n    = n0 + nRow;

  float cv   = (float)cnt[n];
  float invc = cv > 0.0f ? 1.0f / cv : 0.0f;

  const float* nfr = nf   + (size_t)n * NODE_IN;
  const float* msr = msum + (size_t)n * HIDDEN;

  // A operand (16 nodes x K=128 as 4 steps); element i -> K = kb+(i<8?i:i+8)
  v16h a[4];
  #pragma unroll
  for (int s = 0; s < 4; ++s) {
    const float* base = (s < 2) ? (nfr + s * 32) : (msr + (s - 2) * 32);
    const float4* r0 = (const float4*)(base + hi * 8);
    const float4* r1 = (const float4*)(base + 16 + hi * 8);
    float4 f0 = r0[0], f1 = r0[1], f2 = r1[0], f3 = r1[1];
    float tmp[16] = { f0.x,f0.y,f0.z,f0.w, f1.x,f1.y,f1.z,f1.w,
                      f2.x,f2.y,f2.z,f2.w, f3.x,f3.y,f3.z,f3.w };
    float scale = (s < 2) ? 1.0f : invc;        // mean-normalize h_neigh
    #pragma unroll
    for (int i = 0; i < 16; ++i) a[s][i] = (_Float16)(tmp[i] * scale);
  }

  #pragma unroll
  for (int nb = 0; nb < 4; ++nb) {
    float bv = bn[nb * 16 + nRow];              // C col = lane%16 -> bias
    v8f c = { bv, bv, bv, bv, bv, bv, bv, bv };
    #pragma unroll
    for (int s = 0; s < 4; ++s) {
      U16h b;
      const float4* wp = (const float4*)(wnswz + (size_t)((nb * 4 + s) * 32 + lane) * 16);
      b.q[0] = wp[0]; b.q[1] = wp[1];
      c = __builtin_amdgcn_wmma_f32_16x16x32_f16(
          false, a[s], false, b.h, (short)0, c, false, false);
    }
    #pragma unroll
    for (int r = 0; r < 8; ++r)
      out[(size_t)(n0 + r + 8 * hi) * HIDDEN + nb * 16 + nRow] = relu1(c[r]);
  }
}

// ---------------------------------------------------------------------------
extern "C" void kernel_launch(void* const* d_in, const int* in_sizes, int n_in,
                              void* d_out, int out_size, void* d_ws, size_t ws_size,
                              hipStream_t stream)
{
  const float* nf  = (const float*)d_in[0];   // node_features (32768,64)
  const float* ef  = (const float*)d_in[1];   // edge_features (65536,32)
  const int*   src = (const int*)  d_in[2];
  const int*   dst = (const int*)  d_in[3];
  const float* We  = (const float*)d_in[4];   // (32,4096)
  const float* be  = (const float*)d_in[5];   // (4096,)
  const float* Wn  = (const float*)d_in[6];   // (128,64)
  const float* bn  = (const float*)d_in[7];   // (64,)
  float* out = (float*)d_out;

  char* ws = (char*)d_ws;                     // ~2.41 MB total
  _Float16* wswz  = (_Float16*)(ws);                            // 262144 B
  _Float16* wnswz = (_Float16*)(ws + 262144);                   //  16384 B
  float*    msum  = (float*)   (ws + 262144 + 16384);           // 2097152 B
  unsigned* cnt   = (unsigned*)(ws + 262144 + 16384 + 2097152); //  32768 B

  zero_kernel<<<(N_DSTC * HIDDEN + 255) / 256, 256, 0, stream>>>(msum, cnt);
  prep_kernel<<<(256 * 32 + 16 * 32 + 255) / 256, 256, 0, stream>>>(We, Wn, wswz, wnswz);
  edge_kernel<<<NEDGE / 128, 256, 0, stream>>>(nf, ef, src, dst, wswz, be, msum, cnt);
  node_kernel<<<N_DSTC / 128, 256, 0, stream>>>(nf, msum, cnt, wnswz, bn, out);
}
